// QueryGNN_30812095381570
// MI455X (gfx1250) — matrix operand order repaired and checked by reading, not compiled, not run on hardware
//
#include <hip/hip_runtime.h>
#include <stdint.h>

typedef __attribute__((ext_vector_type(16))) __bf16 v16bf;
typedef __attribute__((ext_vector_type(8)))  float  v8f;
typedef __attribute__((ext_vector_type(4)))  unsigned v4u;
typedef __attribute__((ext_vector_type(8)))  int      v8i;
typedef __attribute__((ext_vector_type(4)))  int      v4i;

#define IN_DIM 384
#define HID    256
#define QDIM   384
#define HHID   128

__device__ __forceinline__ unsigned short f2bf(float f) {
    union { float f; unsigned u; } v; v.f = f;
    unsigned u = v.u;
    unsigned r = u + 0x7FFFu + ((u >> 16) & 1u);   // round-to-nearest-even
    return (unsigned short)(r >> 16);
}
__device__ __forceinline__ float bf2f(unsigned short h) {
    union { unsigned u; float f; } v; v.u = ((unsigned)h) << 16;
    return v.f;
}
__device__ __forceinline__ unsigned lds_off(const void* p) {
    // flat LDS addresses: low 32 bits are the wave-relative LDS byte offset
    return (unsigned)(unsigned long long)p;
}

// ---------------------------------------------------------------------------
// TDM: DMA a 2D tile [tileD1 rows x tileD0 elems] of bf16 (data_size=2B) from
// a row-major tensor (row stride = stride0 elems) into LDS, inserting 8 DWORDs
// (16 shorts) of padding after every 16 DWORDs (32 shorts) -> LDS row stride
// of 48 shorts. Descriptor bit layout per CDNA5 ISA 8.3/8.4.
// amdgpu-toolchain builtin: 6 args (g0,g1,g2,g3,g4[VADDR4, unused],cpol).
// ---------------------------------------------------------------------------
__device__ __forceinline__ void tdm_load_2d_bf16(
    unsigned ldsOff, unsigned long long gaddr,
    unsigned tensorD0, unsigned tensorD1, unsigned long long stride0,
    unsigned tileD0, unsigned tileD1)
{
    v4u g0;
    g0[0] = 1u;                                            // count=1, user mode
    g0[1] = ldsOff;                                        // lds_addr (bytes)
    g0[2] = (unsigned)(gaddr & 0xFFFFFFFFull);             // global_addr lo
    g0[3] = (unsigned)((gaddr >> 32) & 0x01FFFFFFull)      // global_addr hi
          | (2u << 30);                                    // type = 2 (image)
    v8i g1;
    g1[0] = (int)((1u << 16)      // data_size = 2B
                | (1u << 20)      // pad_enable
                | (3u << 22)      // pad_interval: 16 DWORDs
                | (7u << 25));    // pad_amount:   8 DWORDs
    g1[1] = (int)((tensorD0 & 0xFFFFu) << 16);                             // dim0 lo
    g1[2] = (int)(((tensorD0 >> 16) & 0xFFFFu) | ((tensorD1 & 0xFFFFu) << 16));
    g1[3] = (int)(((tensorD1 >> 16) & 0xFFFFu) | (tileD0 << 16));
    g1[4] = (int)(tileD1 & 0xFFFFu);                                        // tile_dim2=0
    g1[5] = (int)(unsigned)(stride0 & 0xFFFFFFFFull);
    g1[6] = (int)((unsigned)((stride0 >> 32) & 0xFFFFu));                   // stride1=0
    g1[7] = 0;
    v4i g2 = {0, 0, 0, 0};
    v4i g3 = {0, 0, 0, 0};
    v8i g4 = {0, 0, 0, 0, 0, 0, 0, 0};   // VADDR4 group: unused, zero
    __builtin_amdgcn_tensor_load_to_lds(g0, g1, g2, g3, g4, 0);
}

// ---------------------------------------------------------------------------
// Tiled bf16 WMMA GEMM:  Out[M,NCOL] = relu(A[M,K] * Bt[NCOL,K]^T + bias)
// 256 threads = 8 waves. Block tile: 64 rows x NCOL cols. K stepped by 32.
// Tile staging: Tensor Data Mover (double-buffered), issued by wave 0,
// synchronized with s_wait_tensorcnt + workgroup barriers.
// Fragment layouts follow CDNA5 ISA 7.12.2 (16-bit A 16x32, B 32x16, f32 C/D).
// ---------------------------------------------------------------------------
template<int K, int NCOL, bool OUT_BF16>
__global__ __launch_bounds__(256) void gemm_bf16_wmma(
    const unsigned short* __restrict__ A,    // [M,K]   bf16 row-major
    const unsigned short* __restrict__ Bt,   // [NCOL,K] bf16 (transposed weight)
    const float* __restrict__ bias,          // [NCOL]  fp32 (q-part folded in)
    void* __restrict__ Out, int M, int outStride)
{
    constexpr int CPW = NCOL / 128;          // 16-col tiles per wave (2 or 1)
    __shared__ unsigned short As[2][64][48];
    __shared__ unsigned short Bs[2][NCOL][48];

    const int tid  = threadIdx.x;
    const int wave = tid >> 5;
    const int lane = tid & 31;
    const int half = lane >> 4;
    const int nl   = lane & 15;
    const int rowBase = blockIdx.x * 64;
    const int rowsLeft = M - rowBase;
    const int tileRows = rowsLeft < 64 ? rowsLeft : 64;

    // M-tail block: TDM clamps tile_dim1, so zero the LDS rows it won't write
    if (tileRows < 64) {
        for (int i = tid; i < 2 * 64 * 6; i += 256) {      // 6 x uint4 per 48-short row
            int b = i / (64 * 6), rem = i % (64 * 6);
            int r = rem / 6, c = rem % 6;
            if (r >= tileRows)
                *(uint4*)(&As[b][r][c * 8]) = make_uint4(0u, 0u, 0u, 0u);
        }
    }

    const unsigned long long Ag = (unsigned long long)A + (unsigned long long)rowBase * K * 2;
    const unsigned long long Bg = (unsigned long long)Bt;

    if (wave == 0) {   // prologue: DMA K-step 0 into buffer 0
        tdm_load_2d_bf16(lds_off(&As[0][0][0]), Ag, K, (unsigned)M, K, 32, (unsigned)tileRows);
        tdm_load_2d_bf16(lds_off(&Bs[0][0][0]), Bg, K, NCOL, K, 32, NCOL);
    }

    v8f zero = {0.f,0.f,0.f,0.f,0.f,0.f,0.f,0.f};
    v8f acc[4][CPW];
    #pragma unroll
    for (int r = 0; r < 4; ++r)
        #pragma unroll
        for (int j = 0; j < CPW; ++j) acc[r][j] = zero;

    union U { v16bf v; uint4 q[2]; };

    int kb = 0;
    for (int kt = 0; kt < K; kt += 32, kb ^= 1) {
        __syncthreads();   // previous compute done -> other buffer is free
        if (wave == 0) {
            if (kt + 32 < K) {   // prefetch next K-step while we compute this one
                tdm_load_2d_bf16(lds_off(&As[kb ^ 1][0][0]), Ag + (kt + 32) * 2,
                                 K, (unsigned)M, K, 32, (unsigned)tileRows);
                tdm_load_2d_bf16(lds_off(&Bs[kb ^ 1][0][0]), Bg + (kt + 32) * 2,
                                 K, NCOL, K, 32, NCOL);
                __builtin_amdgcn_s_wait_tensorcnt(2);   // oldest pair (current buf) done
            } else {
                __builtin_amdgcn_s_wait_tensorcnt(0);
            }
        }
        __syncthreads();   // buffer kb visible to all waves

        // A fragments: lane<16 holds M=nl, K{0..7,16..23}; lane>=16 K{8..15,24..31}
        v16bf a[4], b[CPW];
        #pragma unroll
        for (int r = 0; r < 4; ++r) {
            U u;
            u.q[0] = *(const uint4*)(&As[kb][r * 16 + nl][half * 8]);
            u.q[1] = *(const uint4*)(&As[kb][r * 16 + nl][16 + half * 8]);
            a[r] = u.v;
        }
        // B fragments: lane<16 holds N=nl, K{0..7,16..23}; lane>=16 K{8..15,24..31}
        #pragma unroll
        for (int j = 0; j < CPW; ++j) {
            int n = (wave * CPW + j) * 16 + nl;
            U u;
            u.q[0] = *(const uint4*)(&Bs[kb][n][half * 8]);
            u.q[1] = *(const uint4*)(&Bs[kb][n][16 + half * 8]);
            b[j] = u.v;
        }
        #pragma unroll
        for (int r = 0; r < 4; ++r)
            #pragma unroll
            for (int j = 0; j < CPW; ++j)
                acc[r][j] = __builtin_amdgcn_wmma_f32_16x16x32_bf16(
                    false, a[r], false, b[j], (short)0, acc[r][j], false, false);
    }

    // epilogue: bias + ReLU, store (C/D layout: vgpr i -> M = i + 8*half)
    if (tileRows == 64) {   // fast path: no row guards
        #pragma unroll
        for (int r = 0; r < 4; ++r) {
            int row0 = rowBase + r * 16 + half * 8;
            #pragma unroll
            for (int j = 0; j < CPW; ++j) {
                int col = (wave * CPW + j) * 16 + nl;
                float bv = bias[col];
                #pragma unroll
                for (int i = 0; i < 8; ++i) {
                    float v = acc[r][j][i] + bv;
                    v = v > 0.f ? v : 0.f;
                    if constexpr (OUT_BF16)
                        ((unsigned short*)Out)[(size_t)(row0 + i) * outStride + col] = f2bf(v);
                    else
                        ((float*)Out)[(size_t)(row0 + i) * outStride + col] = v;
                }
            }
        }
    } else {
        #pragma unroll
        for (int r = 0; r < 4; ++r) {
            int row0 = rowBase + r * 16 + half * 8;
            #pragma unroll
            for (int j = 0; j < CPW; ++j) {
                int col = (wave * CPW + j) * 16 + nl;
                float bv = bias[col];
                #pragma unroll
                for (int i = 0; i < 8; ++i) {
                    int row = row0 + i;
                    if (row < M) {
                        float v = acc[r][j][i] + bv;
                        v = v > 0.f ? v : 0.f;
                        if constexpr (OUT_BF16)
                            ((unsigned short*)Out)[(size_t)row * outStride + col] = f2bf(v);
                        else
                            ((float*)Out)[(size_t)row * outStride + col] = v;
                    }
                }
            }
        }
    }
}

// --------------------------- graph / scatter kernels -----------------------
__global__ void count_deg(const int* __restrict__ dst, float* __restrict__ deg, int E) {
    int t = blockIdx.x * 256 + threadIdx.x;
    if (t < E) atomicAdd(&deg[dst[t]], 1.0f);
}

template<int D>
__global__ void scatter_add_f32(const float* __restrict__ x, const int* __restrict__ src,
                                const int* __restrict__ dst, float* __restrict__ agg, int E) {
    long long t = (long long)blockIdx.x * 256 + threadIdx.x;
    if (t >= (long long)E * D) return;
    int e = (int)(t / D), f = (int)(t % D);
    atomicAdd(&agg[(size_t)dst[e] * D + f], x[(size_t)src[e] * D + f]);
}

template<int D, int STRIDE>
__global__ void scatter_add_bf16(const unsigned short* __restrict__ h, const int* __restrict__ src,
                                 const int* __restrict__ dst, float* __restrict__ agg, int E) {
    long long t = (long long)blockIdx.x * 256 + threadIdx.x;
    if (t >= (long long)E * D) return;
    int e = (int)(t / D), f = (int)(t % D);
    atomicAdd(&agg[(size_t)dst[e] * D + f], bf2f(h[(size_t)src[e] * STRIDE + f]));
}

// A1[row] = [ bf16(x) | bf16((agg1 + x) / (deg+1)) ]
__global__ void build_A1(const float* __restrict__ x, const float* __restrict__ agg,
                         const float* __restrict__ deg, unsigned short* __restrict__ A1, int Nn) {
    long long t = (long long)blockIdx.x * 256 + threadIdx.x;
    if (t >= (long long)Nn * IN_DIM) return;
    int ti = (int)t;
    int row = ti / IN_DIM, f = ti % IN_DIM;
    float xv  = x[t];
    float inv = 1.0f / (deg[row] + 1.0f);
    A1[(size_t)row * 768 + f]          = f2bf(xv);
    A1[(size_t)row * 768 + IN_DIM + f] = f2bf((agg[t] + xv) * inv);
}

// A2[row, 256+f] = bf16((agg2 + h1) / (deg+1));  h1 = A2[row, f] (bf16, from GEMM1)
__global__ void build_A2(const float* __restrict__ agg, const float* __restrict__ deg,
                         unsigned short* __restrict__ A2, int Nn) {
    long long t = (long long)blockIdx.x * 256 + threadIdx.x;
    if (t >= (long long)Nn * HID) return;
    int ti = (int)t;
    int row = ti >> 8, f = ti & 255;
    float hv  = bf2f(A2[(size_t)row * 512 + f]);
    float inv = 1.0f / (deg[row] + 1.0f);
    A2[(size_t)row * 512 + HID + f] = f2bf((agg[t] + hv) * inv);
}

// Wt[n,k] = bf16(W[k,n]) for the first K rows of W
template<int NCOL>
__global__ void transpose_cvt(const float* __restrict__ W, unsigned short* __restrict__ Wt, int K) {
    int t = blockIdx.x * 256 + threadIdx.x;
    if (t >= K * NCOL) return;
    int k = t / NCOL, n = t % NCOL;    // NCOL is a power of two -> shifts
    Wt[(size_t)n * K + k] = f2bf(W[t]);
}

// c[n] = b[n] + sum_k q[k] * W[kOff+k, n]   (fold row-invariant query term into bias)
__global__ void fold_bias(const float* __restrict__ W, const float* __restrict__ q,
                          const float* __restrict__ b, float* __restrict__ c,
                          int kOff, int kLen, int NCOL) {
    int n = threadIdx.x;
    if (n >= NCOL) return;
    float acc = b[n];
    for (int k = 0; k < kLen; ++k)
        acc += q[k] * W[(size_t)(kOff + k) * NCOL + n];
    c[n] = acc;
}

// logits[row] = s[row,:] . wh2 + bh2  (one wave32 per row, shuffle reduction)
__global__ void head_out(const float* __restrict__ s, const float* __restrict__ wh2,
                         const float* __restrict__ bh2, float* __restrict__ out, int M) {
    int row  = blockIdx.x * 8 + (threadIdx.x >> 5);
    int lane = threadIdx.x & 31;
    if (row >= M) return;
    float acc = 0.f;
    #pragma unroll
    for (int i = 0; i < 4; ++i)
        acc += s[(size_t)row * HHID + lane + i * 32] * wh2[lane + i * 32];
    #pragma unroll
    for (int off = 16; off > 0; off >>= 1)
        acc += __shfl_down(acc, off, 32);
    if (lane == 0) out[row] = acc + bh2[0];
}

// ---------------------------------------------------------------------------
extern "C" void kernel_launch(void* const* d_in, const int* in_sizes, int n_in,
                              void* d_out, int out_size, void* d_ws, size_t ws_size,
                              hipStream_t stream) {
    const float* x   = (const float*)d_in[0];
    const int*   ei  = (const int*)  d_in[1];
    const float* q   = (const float*)d_in[2];
    const float* W1  = (const float*)d_in[3];
    const float* b1  = (const float*)d_in[4];
    const float* W2  = (const float*)d_in[5];
    const float* b2  = (const float*)d_in[6];
    const float* Wh1 = (const float*)d_in[7];
    const float* bh1 = (const float*)d_in[8];
    const float* Wh2 = (const float*)d_in[9];
    const float* bh2 = (const float*)d_in[10];
    float* out = (float*)d_out;

    const int Nn = in_sizes[0] / IN_DIM;      // 50000
    const int E  = in_sizes[1] / 2;           // 400000
    const int* src = ei;
    const int* dst = ei + E;

    // ---- workspace carving (regions aliased across pipeline stages) ----
    char* ws = (char*)d_ws;
    size_t off = 0;
    auto carve = [&](size_t bytes) -> void* {
        void* p = ws + off;
        off += (bytes + 255) & ~(size_t)255;
        return p;
    };
    float* deg = (float*)carve((size_t)Nn * 4);
    char* R1 = (char*)carve((size_t)Nn * IN_DIM * 4);   // 76.8 MB
    char* R2 = (char*)carve((size_t)Nn * 768 * 2);      // 76.8 MB
    unsigned short* Wt1 = (unsigned short*)carve(768 * 256 * 2);
    unsigned short* Wt2 = (unsigned short*)carve(512 * 256 * 2);
    unsigned short* Wth = (unsigned short*)carve(256 * 128 * 2);
    float* c1 = (float*)carve(256 * 4);
    float* c2 = (float*)carve(256 * 4);
    float* ch = (float*)carve(128 * 4);

    // lifetime-aliased views
    float*          agg1 = (float*)R1;            // [N,384] live: steps 1-4
    unsigned short* A2b  = (unsigned short*)R1;   // [N,512] live: GEMM1 -> GEMM2
    float*          sbuf = (float*)R1;            // [N,128] live: GEMM3 -> head
    unsigned short* A1b  = (unsigned short*)R2;   // [N,768] live: build -> GEMM1
    float*          agg2 = (float*)R2;            // [N,256] live: scatter2 -> build_A2
    unsigned short* A3b  = (unsigned short*)R2;   // [N,256] live: GEMM2 -> GEMM3

    // ---- layer 1 aggregation ----
    (void)hipMemsetAsync(deg, 0, (size_t)Nn * 4, stream);
    (void)hipMemsetAsync(agg1, 0, (size_t)Nn * IN_DIM * 4, stream);
    count_deg<<<(E + 255) / 256, 256, 0, stream>>>(dst, deg, E);
    {
        long long tot = (long long)E * IN_DIM;
        scatter_add_f32<IN_DIM><<<(unsigned)((tot + 255) / 256), 256, 0, stream>>>(x, src, dst, agg1, E);
    }
    build_A1<<<(unsigned)(((long long)Nn * IN_DIM + 255) / 256), 256, 0, stream>>>(x, agg1, deg, A1b, Nn);

    // ---- weight prep (independent; any time before GEMM1) ----
    transpose_cvt<256><<<(768 * 256 + 255) / 256, 256, 0, stream>>>(W1, Wt1, 768);
    transpose_cvt<256><<<(512 * 256 + 255) / 256, 256, 0, stream>>>(W2, Wt2, 512);
    transpose_cvt<128><<<(256 * 128 + 255) / 256, 256, 0, stream>>>(Wh1, Wth, 256);
    fold_bias<<<1, 256, 0, stream>>>(W1, q, b1, c1, 768, QDIM, 256);
    fold_bias<<<1, 256, 0, stream>>>(W2, q, b2, c2, 512, QDIM, 256);
    fold_bias<<<1, 128, 0, stream>>>(Wh1, q, bh1, ch, 256, QDIM, 128);

    const int gm = (Nn + 63) / 64;

    // ---- layer 1 GEMM: h1 = relu(A1 @ Wt1^T + c1) -> bf16 into A2[:, :256] ----
    gemm_bf16_wmma<768, 256, true><<<gm, 256, 0, stream>>>(A1b, Wt1, c1, A2b, Nn, 512);

    // ---- layer 2 aggregation (A1 region now dead -> agg2 reuses it) ----
    (void)hipMemsetAsync(agg2, 0, (size_t)Nn * HID * 4, stream);
    {
        long long tot = (long long)E * HID;
        scatter_add_bf16<HID, 512><<<(unsigned)((tot + 255) / 256), 256, 0, stream>>>(A2b, src, dst, agg2, E);
    }
    build_A2<<<(unsigned)(((long long)Nn * HID + 255) / 256), 256, 0, stream>>>(agg2, deg, A2b, Nn);

    // ---- layer 2 GEMM: h2 = relu(A2 @ Wt2^T + c2) -> bf16 A3 ----
    gemm_bf16_wmma<512, 256, true><<<gm, 256, 0, stream>>>(A2b, Wt2, c2, A3b, Nn, 256);

    // ---- head GEMM: s = relu(A3 @ Wth^T + ch) -> fp32 ----
    gemm_bf16_wmma<256, 128, false><<<gm, 256, 0, stream>>>(A3b, Wth, ch, sbuf, Nn, 128);

    // ---- final 128-dot + bias ----
    head_out<<<(Nn + 7) / 8, 256, 0, stream>>>(sbuf, Wh2, bh2, out, Nn);
}
